// cheb_conv_aagcn_5377299055095
// MI455X (gfx1250) — matrix operand order repaired and checked by reading, not compiled, not run on hardware
//
#include <hip/hip_runtime.h>
#include <hip/hip_bf16.h>
#include <math.h>

// Problem constants (from reference)
#define NN   2048
#define BB   8
#define KK   3
#define F_IN 16
#define F_OUT 64
#define TT   12
#define FT   (F_IN * TT)   // 192

typedef __attribute__((ext_vector_type(16))) __bf16 v16bf;
typedef __attribute__((ext_vector_type(8)))  float  v8f;
typedef __attribute__((ext_vector_type(4))) unsigned int u32x4;
typedef __attribute__((ext_vector_type(8))) int i32x8;
typedef __attribute__((ext_vector_type(4))) int i32x4;

// Native bf16 conversion (gfx1250 has hardware bf16 cvt)
static __device__ __forceinline__ unsigned short f32_to_bf16(float f) {
    union { __bf16 h; unsigned short u; } c;
    c.h = (__bf16)f;
    return c.u;
}

// ---------------------------------------------------------------------------
// TDM: load one B tile (192 ft-rows x 32 m of bf16 = 64 B/row) from xt[b] into
// LDS with hardware padding to 80 B rows (pad 16 B every 64 B stored).
// D# per CDNA5 ISA 8.3-8.6. 2D tensor: dim0 = m (2048), dim1 = ft (192).
// ---------------------------------------------------------------------------
static __device__ __forceinline__ void tdm_load_tile(const unsigned short* gsrc,
                                                     unsigned ldsoff) {
    unsigned long long ga = (unsigned long long)(uintptr_t)gsrc;
    u32x4 g0 = { 1u,                                   // count=1, load, normal
                 ldsoff,                               // lds_addr
                 (unsigned)(ga & 0xFFFFFFFFu),         // global_addr[31:0]
                 (unsigned)((ga >> 32) & 0x01FFFFFFu) | 0x80000000u }; // [56:32] | type=2
    i32x8 g1 = { (int)((1u << 16)      // data_size = 2 bytes
                     | (1u << 20)      // pad_enable
                     | (3u << 22)      // pad_interval: 16 DWORDs = 64 B
                     | (3u << 25)),    // pad_amount:   4 DWORDs = 16 B
                 (int)(0x0800u << 16), // tensor_dim0 = 2048 (low16 -> bits 31:16)
                 (int)(192u << 16),    // dim0 hi = 0 | tensor_dim1 = 192 (low16)
                 (int)(32u << 16),     // dim1 hi = 0 | tile_dim0 = 32
                 (int)(192u),          // tile_dim1 = 192 | tile_dim2 = 0
                 (int)(2048),          // tensor_dim0_stride[31:0] = 2048 elements
                 0, 0 };
    i32x4 gz = { 0, 0, 0, 0 };
#if __clang_major__ >= 23
    i32x8 gz8 = { 0, 0, 0, 0, 0, 0, 0, 0 };
    __builtin_amdgcn_tensor_load_to_lds(g0, g1, gz, gz, gz8, 0);
#else
    __builtin_amdgcn_tensor_load_to_lds(g0, g1, gz, gz, 0);
#endif
}

// ---------------------------------------------------------------------------
// 1) deg[n] = sum_m adj[n][m]
// ---------------------------------------------------------------------------
__global__ __launch_bounds__(256) void deg_kernel(const float* __restrict__ adj,
                                                  float* __restrict__ deg) {
    __shared__ float red[256];
    int row = blockIdx.x;
    float s = 0.f;
    for (int m = threadIdx.x; m < NN; m += 256) s += adj[(size_t)row * NN + m];
    red[threadIdx.x] = s;
    __syncthreads();
    for (int off = 128; off > 0; off >>= 1) {
        if (threadIdx.x < off) red[threadIdx.x] += red[threadIdx.x + off];
        __syncthreads();
    }
    if (threadIdx.x == 0) deg[row] = red[0];
}

// ---------------------------------------------------------------------------
// 2) per-(b,k,n) online softmax stats over m: max and 1/sum(exp)
// ---------------------------------------------------------------------------
__global__ __launch_bounds__(256) void stats_kernel(const float* __restrict__ sa,
                                                    float* __restrict__ mx,
                                                    float* __restrict__ rinv) {
    int n  = blockIdx.x * 256 + threadIdx.x;
    int bk = blockIdx.y;
    const float* p = sa + (size_t)bk * NN * NN;
    float m = -INFINITY, s = 0.f;
    for (int i = 0; i < NN; ++i) {
        float v  = p[(size_t)i * NN + n];
        float nm = fmaxf(m, v);
        s = s * __expf(m - nm) + __expf(v - nm);
        m = nm;
    }
    mx[(size_t)bk * NN + n]   = m;
    rinv[(size_t)bk * NN + n] = 1.0f / s;
}

// ---------------------------------------------------------------------------
// 3) xt[b][ft][m] = bf16(x[b][m][ft])   (transposed, bf16; 6.3 MB, L2-hot)
// ---------------------------------------------------------------------------
__global__ __launch_bounds__(256) void xt_kernel(const float* __restrict__ x,
                                                 unsigned short* __restrict__ xt) {
    int i  = blockIdx.x * 256 + threadIdx.x;   // over B*FT*NN
    int m  = i & (NN - 1);
    int ft = (i >> 11) % FT;
    int b  = i / (NN * FT);
    float v = x[((size_t)b * NN + m) * FT + ft];
    xt[i] = f32_to_bf16(v);
}

// ---------------------------------------------------------------------------
// 4) Fused softmax+cheb+GEMM for k=1,2 (bf16 WMMA, f32 accumulate)
//    C[b,kk,n,ft] = sum_m cheb[k][m,n]*attn[b,k,m,n] * x[b,m,ft]
//    Block: 256 threads = 8 waves; wave owns 16 n-rows x 192 ft columns.
//    B tiles double-buffered in LDS, filled by the Tensor Data Mover.
// ---------------------------------------------------------------------------
__global__ __launch_bounds__(256) void gemm_kernel(
    const float* __restrict__ sa,  const float* __restrict__ adj,
    const float* __restrict__ deg, const float* __restrict__ mx,
    const float* __restrict__ rinv, const unsigned short* __restrict__ xt,
    float* __restrict__ C) {
    // two B tiles: 192 ft-rows x 32 m, rows padded to 40 ushorts (80 B) by TDM
    __shared__ __align__(16) unsigned short Xs[2 * FT * 40];

    int b    = blockIdx.y >> 1;
    int kk   = blockIdx.y & 1;          // 0 -> cheb k=1 (L), 1 -> k=2 (2L^2 - I)
    int k    = kk + 1;
    int tid  = threadIdx.x;
    int wave = tid >> 5, lane = tid & 31;
    int n0   = blockIdx.x * 128 + wave * 16;
    int row  = lane & 15;
    int n    = n0 + row;
    int kbA  = (lane < 16) ? 0 : 16;    // A-frag K base (contiguous k = 16h + s)
    int kbB  = (lane < 16) ? 0 : 16;    // B-frag K base (contiguous k = 16h + s, matches A)

    const float* saB = sa + ((size_t)(b * KK + k)) * NN * NN;
    const unsigned short* xb = xt + (size_t)b * FT * NN;
    float mxv = mx[(size_t)(b * KK + k) * NN + n];
    float rsv = rinv[(size_t)(b * KK + k) * NN + n];
    unsigned ldsXs = (unsigned)(uintptr_t)&Xs[0];

    v8f c[12] = {};

    // prologue: DMA first tile into buffer 0 (wave 0 only; TDM ignores EXEC)
    if (tid < 32) tdm_load_tile(xb, ldsXs);

    for (int it = 0; it < 64; ++it) {
        int m0 = it * 32;

        // --- build A fragment on the fly (overlaps the TDM transfer) ---
        union AU { v16bf v; __bf16 h[16]; } a;
        #pragma unroll
        for (int e = 0; e < 16; ++e) {
            int ml = kbA + e;  // contiguous k order, matches B fragment
            int m  = m0 + ml;
            float sv = saB[(size_t)m * NN + n];
            float av = adj[(size_t)m * NN + n];
            float dg = (m == n) ? 1.0f : 0.0f;
            float L  = dg * deg[m] - av;
            float ch = (kk == 0) ? L : (2.0f * L * L - dg);
            a.h[e] = (__bf16)(__expf(sv - mxv) * rsv * ch);
        }

        // --- wave 0: issue next tile's DMA, then wait for current tile ---
        if (tid < 32) {
            if (it < 63) {
                tdm_load_tile(xb + (size_t)(m0 + 32),
                              ldsXs + (unsigned)(((it + 1) & 1) * (FT * 80)));
                __builtin_amdgcn_s_wait_tensorcnt(1);
            } else {
                __builtin_amdgcn_s_wait_tensorcnt(0);
            }
        }
        __syncthreads();   // current tile visible to all waves

        // --- 12 WMMAs: 16n x 192ft x 32m, B frags double-buffered ---
        const unsigned short* Xb = &Xs[(it & 1) * (FT * 40)];
        union BU { v16bf v; uint4 q[2]; } bu[2];
        {
            const unsigned short* base = Xb + (size_t)row * 40 + kbB;
            bu[0].q[0] = *(const uint4*)(base);
            bu[0].q[1] = *(const uint4*)(base + 8);
        }
        #pragma unroll
        for (int j = 0; j < 12; ++j) {
            if (j < 11) {
                const unsigned short* base =
                    Xb + (size_t)((j + 1) * 16 + row) * 40 + kbB;
                bu[(j + 1) & 1].q[0] = *(const uint4*)(base);
                bu[(j + 1) & 1].q[1] = *(const uint4*)(base + 8);
            }
            c[j] = __builtin_amdgcn_wmma_f32_16x16x32_bf16(
                       false, a.v, false, bu[j & 1].v, (short)0, c[j], false, false);
        }
        __syncthreads();   // everyone done reading this buffer; safe to refill
    }

    // --- store C tile (ISA C/D layout: VGPR r -> M=r / M=r+8 by half-wave) ---
    float* Cb = C + ((size_t)(b * 2 + kk) * NN) * FT;
    #pragma unroll
    for (int j = 0; j < 12; ++j) {
        int ft = j * 16 + row;
        #pragma unroll
        for (int r = 0; r < 8; ++r) {
            int nOut = n0 + ((lane < 16) ? r : (r + 8));
            Cb[(size_t)nOut * FT + ft] = c[j][r];
        }
    }
}

// ---------------------------------------------------------------------------
// 5) Epilogue: k=0 diagonal term + Theta contraction + ReLU
//    out[b,n,o,t] = relu( sum_f Th0[f,o]*d[n]*x[b,n,f,t]
//                        + Th1[f,o]*C1[b,n,f,t] + Th2[f,o]*C2[b,n,f,t] )
//    with d[n] = attn[b,0,n,n] = exp(sa0[n,n]-mx0)/sum0.
// ---------------------------------------------------------------------------
__global__ __launch_bounds__(256) void out_kernel(
    const float* __restrict__ sa, const float* __restrict__ x,
    const float* __restrict__ Theta, const float* __restrict__ mx,
    const float* __restrict__ rinv, const float* __restrict__ C,
    float* __restrict__ out) {
    __shared__ float C1s[8 * FT], C2s[8 * FT], Xss[8 * FT], ds[8];
    int gid = blockIdx.x * 8;           // row-group over B*NN
    int b   = gid / NN;
    int nb  = gid % NN;
    int tid = threadIdx.x;

    for (int i = tid; i < 8 * FT; i += 256) {
        int r = i / FT, ft = i % FT;
        int n = nb + r;
        C1s[i] = C[((size_t)(b * 2 + 0) * NN + n) * FT + ft];
        C2s[i] = C[((size_t)(b * 2 + 1) * NN + n) * FT + ft];
        Xss[i] = x[((size_t)b * NN + n) * FT + ft];
    }
    if (tid < 8) {
        int n = nb + tid;
        float m0 = mx[(size_t)(b * KK) * NN + n];
        float r0 = rinv[(size_t)(b * KK) * NN + n];
        float sv = sa[((size_t)(b * KK) * NN + n) * NN + n];
        ds[tid] = __expf(sv - m0) * r0;
    }
    __syncthreads();

    for (int idx = tid; idx < 8 * F_OUT * TT; idx += 256) {
        int r   = idx / (F_OUT * TT);
        int rem = idx % (F_OUT * TT);
        int o = rem / TT, t = rem % TT;
        float d = ds[r];
        float acc = 0.f;
        #pragma unroll
        for (int f = 0; f < F_IN; ++f) {
            float th0 = Theta[(0 * F_IN + f) * F_OUT + o];
            float th1 = Theta[(1 * F_IN + f) * F_OUT + o];
            float th2 = Theta[(2 * F_IN + f) * F_OUT + o];
            int ci = r * FT + f * TT + t;
            acc += th0 * d * Xss[ci] + th1 * C1s[ci] + th2 * C2s[ci];
        }
        int n = nb + r;
        out[(((size_t)b * NN + n) * F_OUT + o) * TT + t] = fmaxf(acc, 0.f);
    }
}

// ---------------------------------------------------------------------------
extern "C" void kernel_launch(void* const* d_in, const int* in_sizes, int n_in,
                              void* d_out, int out_size, void* d_ws, size_t ws_size,
                              hipStream_t stream) {
    (void)in_sizes; (void)n_in; (void)out_size; (void)ws_size;
    const float* x     = (const float*)d_in[0];   // (B,N,F_IN,T)
    const float* sa    = (const float*)d_in[1];   // (B,K,N,N)
    const float* adj   = (const float*)d_in[2];   // (N,N)
    const float* Theta = (const float*)d_in[3];   // (K,F_IN,F_OUT)

    char* ws = (char*)d_ws;
    float*          deg  = (float*)(ws + 0);                       //   8 KB
    float*          mxp  = (float*)(ws + 8192);                    // 384 KB
    float*          rinv = (float*)(ws + 8192 + 196608);
    unsigned short* xt   = (unsigned short*)(ws + 401408);         // 6.3 MB
    float*          C    = (float*)(ws + 6692864);                 // 25.2 MB
    float*          out  = (float*)d_out;

    deg_kernel  <<<NN, 256, 0, stream>>>(adj, deg);
    stats_kernel<<<dim3(NN / 256, BB * KK), 256, 0, stream>>>(sa, mxp, rinv);
    xt_kernel   <<<(BB * FT * NN) / 256, 256, 0, stream>>>(x, xt);
    gemm_kernel <<<dim3(NN / 128, BB * 2), 256, 0, stream>>>(sa, adj, deg, mxp,
                                                             rinv, xt, C);
    out_kernel  <<<(BB * NN) / 8, 256, 0, stream>>>(sa, x, Theta, mxp, rinv, C, out);
}
